// PointNetFeatureUpsampling_78932908966299
// MI455X (gfx1250) — matrix operand order repaired and checked by reading, not compiled, not run on hardware
//
#include <hip/hip_runtime.h>

// ---------------- problem constants (from reference) ----------------
#define BB   16
#define NN   4096
#define SS   512
#define DD   384            // D2
#define KK   5
#define MM   (BB * NN)      // 65536 rows
#define KPAD0 416           // (DD+3)=387 padded up to 13 k-tiles of 32
#define KPAD1 384           // DD, already 12 k-tiles of 32
#define BIGF 1e10f
#define EPSF 1.1920928955078125e-07f   // float32 eps

typedef __attribute__((ext_vector_type(16))) __bf16 v16bf;
typedef __attribute__((ext_vector_type(8)))  __bf16 v8bf;
typedef __attribute__((ext_vector_type(8)))  float  v8f;
typedef __attribute__((ext_vector_type(4)))  unsigned int uint32x4;
typedef __attribute__((ext_vector_type(8)))  int          int32x8;
typedef __attribute__((ext_vector_type(4)))  int          int32x4;

// =====================================================================
// Kernel 1: KNN over ragged keys + inverse-distance weights.
// One thread per query point; xyz2[b] staged in LDS (6 KB).
// =====================================================================
__global__ __launch_bounds__(256) void knn_kernel(
    const float* __restrict__ xyz1, const float* __restrict__ xyz2,
    const int* __restrict__ emb_lens,
    int* __restrict__ idx_out, float* __restrict__ w_out) {
  __shared__ float s_xyz[SS * 3];
  const int b = blockIdx.x;
  const int n = blockIdx.y * 256 + threadIdx.x;

  for (int i = threadIdx.x; i < SS * 3; i += 256)
    s_xyz[i] = xyz2[(size_t)b * SS * 3 + i];
  __syncthreads();

  const int ev = emb_lens[b];
  const size_t q = (size_t)b * NN + n;
  const float qx = xyz1[q * 3 + 0];
  const float qy = xyz1[q * 3 + 1];
  const float qz = xyz1[q * 3 + 2];

  float bd[KK]; int bi[KK];
#pragma unroll
  for (int k = 0; k < KK; ++k) { bd[k] = BIGF; bi[k] = 0; }

  for (int s = 0; s < SS; ++s) {
    const float dx = qx - s_xyz[s * 3 + 0];
    const float dy = qy - s_xyz[s * 3 + 1];
    const float dz = qz - s_xyz[s * 3 + 2];
    float d = dx * dx + dy * dy + dz * dz;
    d = (s < ev) ? d : BIGF;
    if (d < bd[KK - 1]) {
#pragma unroll
      for (int k = 0; k < KK; ++k) {
        if (d < bd[k]) {
#pragma unroll
          for (int j = KK - 1; j > k; --j) { bd[j] = bd[j - 1]; bi[j] = bi[j - 1]; }
          bd[k] = d; bi[k] = s;
          break;
        }
      }
    }
  }

  float rc[KK]; float rsum = 0.0f;
#pragma unroll
  for (int k = 0; k < KK; ++k) { rc[k] = 1.0f / (bd[k] + EPSF); rsum += rc[k]; }
  const float inv = 1.0f / rsum;
  const size_t base = q * KK;
#pragma unroll
  for (int k = 0; k < KK; ++k) {
    idx_out[base + k] = bi[k];
    w_out[base + k]   = rc[k] * inv;
  }
}

// =====================================================================
// Kernel 2: gather + weighted interp + concat coords, pack to bf16
// activation matrix A [MM, KPAD0]. One wave (32 lanes) per point.
// =====================================================================
__global__ __launch_bounds__(256) void interp_pack_kernel(
    const float* __restrict__ points1, const float* __restrict__ points2,
    const int* __restrict__ idx, const float* __restrict__ w,
    __bf16* __restrict__ Abuf) {
  const int lane = threadIdx.x & 31;
  const int wave = threadIdx.x >> 5;
  const size_t p = (size_t)blockIdx.x * 8 + wave;   // point id in [0, MM)
  const int b = (int)(p >> 12);                     // p / NN

  float wk[KK]; int ik[KK];
#pragma unroll
  for (int k = 0; k < KK; ++k) { wk[k] = w[p * KK + k]; ik[k] = idx[p * KK + k]; }

  const float* p2 = points2 + (size_t)b * SS * DD;
  __bf16* arow = Abuf + p * KPAD0;

  for (int c = lane; c < KPAD0; c += 32) {
    float v;
    if (c < 3) {
      v = points1[p * 3 + c];
    } else if (c < 3 + DD) {
      const int cc = c - 3;
      v = 0.0f;
#pragma unroll
      for (int k = 0; k < KK; ++k) v += wk[k] * p2[(size_t)ik[k] * DD + cc];
    } else {
      v = 0.0f;
    }
    arow[c] = (__bf16)v;
  }
}

// =====================================================================
// Weight pack kernels: f32 -> bf16 (W0 zero-padded 387 -> 416 in K).
// =====================================================================
__global__ __launch_bounds__(256) void pack_w0_kernel(const float* __restrict__ W0,
                                                      __bf16* __restrict__ W0bf) {
  const int o = blockIdx.x;                          // 0..383
  for (int c = threadIdx.x; c < KPAD0; c += 256)
    W0bf[(size_t)o * KPAD0 + c] =
        (c < DD + 3) ? (__bf16)W0[(size_t)o * (DD + 3) + c] : (__bf16)0.0f;
}

__global__ __launch_bounds__(256) void pack_w1_kernel(const float* __restrict__ W1,
                                                      __bf16* __restrict__ W1bf) {
  const size_t i = (size_t)blockIdx.x * 256 + threadIdx.x;  // DD*DD
  if (i < (size_t)DD * DD) W1bf[i] = (__bf16)W1[i];
}

// =====================================================================
// Kernel 3/5: bf16 WMMA GEMM  C[M,N] = A[M,Kpad] * W[N,Kpad]^T  (f32 acc)
//
// Block = 256 threads = 8 waves laid 4(M) x 2(N); block tile 64x64.
// The block's 64-row weight slab (64*KPAD bf16, contiguous in memory) is
// staged into LDS once via the Tensor Data Mover (tensor_load_to_lds +
// s_wait_tensorcnt), then B fragments come from LDS (ds_load_b128) —
// removing the 4x redundant per-M-wave weight reads. A fragments stream
// from global with prefetch hints. Fragment layouts per
// CDNA5 ISA §7.12.2 (wave32).
// =====================================================================
template <int KPAD>
__global__ __launch_bounds__(256) void gemm_bf16_wmma_kernel(
    const __bf16* __restrict__ A, const __bf16* __restrict__ Bw,
    float* __restrict__ C, int N) {
  __shared__ __bf16 sB[64 * KPAD];

  const int lane = threadIdx.x & 31;
  const int wave = threadIdx.x >> 5;
  const int waveM = wave & 3;
  const int waveN = wave >> 2;
  const int m0 = blockIdx.x * 64 + waveM * 16;
  const int n0l = waveN * 32;                       // N offset inside block
  const int n0 = blockIdx.y * 64 + n0l;             // global N offset
  const int half = lane >> 4;      // 0 or 1
  const int lid  = lane & 15;

  // ---- stage the 64 x KPAD weight slab into LDS --------------------
  const __bf16* wslab = Bw + (size_t)blockIdx.y * 64 * KPAD;  // contiguous
#if __has_builtin(__builtin_amdgcn_tensor_load_to_lds) && \
    __has_builtin(__builtin_amdgcn_s_wait_tensorcnt)
  if (wave == 0) {
    // Build D# per CDNA5 ISA:
    // 1-D copy, data_size=8B, tensor_dim0 = tile_dim0 = slab bytes / 8.
    const unsigned long long gaddr = (unsigned long long)(uintptr_t)wslab;
    const unsigned int laddr = (unsigned int)(uintptr_t)(&sB[0]);
    const unsigned int nUnits = (64u * (unsigned)KPAD * 2u) >> 3;  // 8B units

    uint32x4 g0 = {};
    g0[0] = 1u;                                     // count=1 (valid), user mode
    g0[1] = laddr;                                  // lds_addr (bytes)
    g0[2] = (unsigned int)(gaddr & 0xFFFFFFFFu);    // global_addr[31:0]
    g0[3] = (unsigned int)((gaddr >> 32) & 0x1FFFFFFu) | (2u << 30); // addr[56:32] | type=2

    int32x8 g1 = {};
    g1[0] = (int)(3u << 16);                        // wg_mask=0 | data_size=3 (8B)
    g1[1] = (int)((nUnits & 0xFFFFu) << 16);        // tensor_dim0[15:0]  @ bits 63:48
    g1[2] = (int)(((nUnits >> 16) & 0xFFFFu)        // tensor_dim0[31:16] @ bits 79:64
                  | (1u << 16));                    // tensor_dim1 = 1    @ bits 111:80
    g1[3] = (int)((nUnits & 0xFFFFu) << 16);        // tile_dim0          @ bits 127:112
    g1[4] = 0;                                      // tile_dim1/2 unused
    g1[5] = (int)nUnits;                            // tensor_dim0_stride[31:0]
    g1[6] = (int)((nUnits & 0xFFFFu) << 16);        // stride0[47:32]=0 | dim1_stride[15:0]
    g1[7] = 0;

    int32x4 gz = {};
#if __clang_major__ >= 23
    int32x8 gz8 = {};
    __builtin_amdgcn_tensor_load_to_lds(g0, g1, gz, gz, gz8, 0);
#else
    __builtin_amdgcn_tensor_load_to_lds(g0, g1, gz, gz, 0);
#endif
    __builtin_amdgcn_s_wait_tensorcnt(0);
  }
#else
  // Fallback: cooperative copy (16B chunks, all 256 threads).
  {
    const int nChunks = (64 * KPAD * 2) / 16;
    const v8bf* src = (const v8bf*)wslab;
    v8bf* dst = (v8bf*)&sB[0];
    for (int i = threadIdx.x; i < nChunks; i += 256) dst[i] = src[i];
  }
#endif
  __syncthreads();

  v8f acc0 = {};
  v8f acc1 = {};

  const __bf16* arow = A + (size_t)(m0 + lid) * KPAD + half * 8;
  const __bf16* lb0  = &sB[(n0l + lid) * KPAD + half * 8];
  const __bf16* lb1  = &sB[(n0l + 16 + lid) * KPAD + half * 8];

  const int Ktiles = KPAD >> 5;
#pragma unroll 4
  for (int kt = 0; kt < Ktiles; ++kt) {
    const int k0 = kt * 32;
    __builtin_prefetch(arow + k0 + 64, 0, 1);        // global_prefetch_b8

    // A fragment: lane(l&15)=row; e<8 -> K=k0+half*8+e ; e>=8 -> +16
    const v8bf alo = *(const v8bf*)(arow + k0);
    const v8bf ahi = *(const v8bf*)(arow + k0 + 16);
    v16bf a;
#pragma unroll
    for (int i = 0; i < 8; ++i) { a[i] = alo[i]; a[8 + i] = ahi[i]; }

    // B fragments from LDS: lane(l&15)=col; e<8 -> K=k0+half*8+e ; e>=8 -> +16
    const v8bf b0lo = *(const v8bf*)(lb0 + k0);
    const v8bf b0hi = *(const v8bf*)(lb0 + k0 + 16);
    const v8bf b1lo = *(const v8bf*)(lb1 + k0);
    const v8bf b1hi = *(const v8bf*)(lb1 + k0 + 16);
    v16bf b0, b1;
#pragma unroll
    for (int i = 0; i < 8; ++i) {
      b0[i] = b0lo[i]; b0[8 + i] = b0hi[i];
      b1[i] = b1lo[i]; b1[8 + i] = b1hi[i];
    }

    acc0 = __builtin_amdgcn_wmma_f32_16x16x32_bf16(false, a, false, b0,
                                                   (short)0, acc0, false, false);
    acc1 = __builtin_amdgcn_wmma_f32_16x16x32_bf16(false, a, false, b1,
                                                   (short)0, acc1, false, false);
  }

  // C layout: elem v -> M = m0 + half*8 + v, N = n0 + (l&15)
#pragma unroll
  for (int v = 0; v < 8; ++v) {
    const size_t r = (size_t)(m0 + half * 8 + v) * N;
    C[r + n0 + lid]      = acc0[v];
    C[r + n0 + 16 + lid] = acc1[v];
  }
}

// =====================================================================
// Kernel 4a/6a: masked per-channel sum / sumsq / count (atomic partials).
// blockDim = 384 (one thread per channel); 256 rows per block, all rows
// of a block share the same batch b (256 | 4096).
// =====================================================================
__global__ __launch_bounds__(384) void bn_stats_kernel(
    const float* __restrict__ X, const int* __restrict__ lens,
    float* __restrict__ sum, float* __restrict__ sumsq, float* __restrict__ cnt) {
  const int c  = threadIdx.x;
  const int r0 = blockIdx.x * 256;
  const int b  = r0 >> 12;
  const int n0 = r0 & (NN - 1);
  const int lim = lens[b];
  float s = 0.0f, qq = 0.0f;
  for (int i = 0; i < 256; ++i) {
    if (n0 + i < lim) {
      const float x = X[(size_t)(r0 + i) * DD + c];
      s += x; qq += x * x;
    }
  }
  atomicAdd(&sum[c], s);
  atomicAdd(&sumsq[c], qq);
  if (c == 0) {
    int m = lim - n0; m = m < 0 ? 0 : (m > 256 ? 256 : m);
    atomicAdd(cnt, (float)m);
  }
}

// =====================================================================
// Kernel 4b/6b: apply masked BN (+ optional exact GELU), write f32
// and/or bf16.
// =====================================================================
__global__ __launch_bounds__(256) void bn_apply_kernel(
    const float* __restrict__ X,
    const float* __restrict__ sum, const float* __restrict__ sumsq,
    const float* __restrict__ cnt,
    const float* __restrict__ gamma, const float* __restrict__ beta,
    const int* __restrict__ lens, int do_gelu,
    float* __restrict__ outF, __bf16* __restrict__ outBF) {
  const size_t i = (size_t)blockIdx.x * 256 + threadIdx.x;   // over MM*DD
  const int c = (int)(i % DD);
  const size_t r = i / DD;
  const int b = (int)(r >> 12);
  const int n = (int)(r & (NN - 1));

  const float ntot = fmaxf(cnt[0], 1.0f);
  const float mean = sum[c] / ntot;
  const float var  = sumsq[c] / ntot - mean * mean;
  const float x  = X[i];
  const float xh = (x - mean) * rsqrtf(var + 1e-5f);
  float y = (xh * gamma[c] + beta[c]) * ((n < lens[b]) ? 1.0f : 0.0f);
  if (do_gelu) y = 0.5f * y * (1.0f + erff(y * 0.70710678118654752f));
  if (outF)  outF[i]  = y;
  if (outBF) outBF[i] = (__bf16)y;
}

// =====================================================================
// Host-side launch
// =====================================================================
extern "C" void kernel_launch(void* const* d_in, const int* in_sizes, int n_in,
                              void* d_out, int out_size, void* d_ws, size_t ws_size,
                              hipStream_t stream) {
  (void)in_sizes; (void)n_in; (void)out_size; (void)ws_size;

  const float* xyz1     = (const float*)d_in[0];
  const float* xyz2     = (const float*)d_in[1];
  const float* points1  = (const float*)d_in[2];
  const float* points2  = (const float*)d_in[3];
  const int*   p_lens   = (const int*)d_in[4];
  const int*   e_lens   = (const int*)d_in[5];
  /* d_in[6] = point_mask (bool) — recomputed from p_lens instead */
  const float* W0       = (const float*)d_in[7];
  const float* gamma0   = (const float*)d_in[8];
  const float* beta0    = (const float*)d_in[9];
  const float* W1       = (const float*)d_in[10];
  const float* gamma1   = (const float*)d_in[11];
  const float* beta1    = (const float*)d_in[12];

  float* out_x  = (float*)d_out;                               // [B,N,DD] f32
  int*   out_ix = (int*)(out_x + (size_t)MM * DD);             // [B,N,KK] i32

  // ---- workspace layout (regions reused across phases) ----
  char* ws = (char*)d_ws;
  size_t off = 0;
  auto carve = [&](size_t bytes) {
    void* p = ws + off;
    off += (bytes + 255) & ~(size_t)255;
    return p;
  };
  float*  wbuf  = (float*)carve((size_t)MM * KK * sizeof(float));
  __bf16* Abuf  = (__bf16*)carve((size_t)MM * KPAD0 * sizeof(__bf16)); // later: X1g
  __bf16* W0bf  = (__bf16*)carve((size_t)DD * KPAD0 * sizeof(__bf16));
  __bf16* W1bf  = (__bf16*)carve((size_t)DD * KPAD1 * sizeof(__bf16));
  float*  X1    = (float*)carve((size_t)MM * DD * sizeof(float));      // later: X2
  float*  stats = (float*)carve(2 * 800 * sizeof(float));
  __bf16* X1g   = Abuf;     // GEMM1 activations dead after GEMM1
  float*  X2    = X1;       // X1 dead after bn_apply stage 1
  float *s1 = stats, *q1 = stats + 384, *c1 = stats + 768;
  float *s2 = stats + 800, *q2 = stats + 1184, *c2 = stats + 1568;

  hipMemsetAsync(stats, 0, 2 * 800 * sizeof(float), stream);

  // 1) KNN + weights (idx straight into output tail)
  knn_kernel<<<dim3(BB, NN / 256), 256, 0, stream>>>(xyz1, xyz2, e_lens, out_ix, wbuf);

  // 2) interpolation + concat -> bf16 activation matrix [MM, 416]
  interp_pack_kernel<<<MM / 8, 256, 0, stream>>>(points1, points2, out_ix, wbuf, Abuf);

  // 3) weight packing
  pack_w0_kernel<<<DD, 256, 0, stream>>>(W0, W0bf);
  pack_w1_kernel<<<(DD * DD + 255) / 256, 256, 0, stream>>>(W1, W1bf);

  // 4) GEMM1 (WMMA bf16, TDM-staged weights): X1 = A @ W0^T
  gemm_bf16_wmma_kernel<KPAD0><<<dim3(MM / 64, DD / 64), 256, 0, stream>>>(Abuf, W0bf, X1, DD);

  // 5) masked BN stats + apply + GELU -> bf16 activations for GEMM2
  bn_stats_kernel<<<MM / 256, 384, 0, stream>>>(X1, p_lens, s1, q1, c1);
  bn_apply_kernel<<<(size_t)MM * DD / 256, 256, 0, stream>>>(
      X1, s1, q1, c1, gamma0, beta0, p_lens, /*gelu=*/1, nullptr, X1g);

  // 6) GEMM2 (WMMA bf16, TDM-staged weights): X2 = X1g @ W1^T
  gemm_bf16_wmma_kernel<KPAD1><<<dim3(MM / 64, DD / 64), 256, 0, stream>>>(X1g, W1bf, X2, DD);

  // 7) masked BN stats + apply -> final f32 output
  bn_stats_kernel<<<MM / 256, 384, 0, stream>>>(X2, p_lens, s2, q2, c2);
  bn_apply_kernel<<<(size_t)MM * DD / 256, 256, 0, stream>>>(
      X2, s2, q2, c2, gamma1, beta1, p_lens, /*gelu=*/0, out_x, nullptr);
}
